// MRCGNN_79276506349793
// MI455X (gfx1250) — matrix operand order repaired and checked by reading, not compiled, not run on hardware
//
#include <hip/hip_runtime.h>
#include <hip/hip_bf16.h>

// ---------------------------------------------------------------------------
// MRCGNN on gfx1250: relation-sorted fused edge transform via bf16 WMMA,
// with per-block async staging of W[r] into LDS (global_load_async_to_lds).
// ---------------------------------------------------------------------------

typedef __attribute__((ext_vector_type(16))) __bf16 v16bf;
typedef __attribute__((ext_vector_type(8)))  __bf16 v8bf;
typedef __attribute__((ext_vector_type(8)))  float  v8f;

#define NNODES 20000
#define NEDGES 640000
#define FDIM   128
#define H1DIM  64
#define H2DIM  32
#define NRELS  65
#define NPAIR  4096
// Each relation's edge list is padded to a multiple of 128 (8 tiles = 1 block)
// so that every workgroup is relation-uniform.
#define EPAD   (NEDGES + 128 * NRELS)

__device__ __forceinline__ v16bf mrc_cat8(v8bf lo, v8bf hi) {
  return __builtin_shufflevector(lo, hi, 0,1,2,3,4,5,6,7,8,9,10,11,12,13,14,15);
}

// --------------------------- small prep kernels ----------------------------

__global__ void mrc_f32_to_bf16(const float* __restrict__ in,
                                __bf16* __restrict__ out, int n) {
  int i = blockIdx.x * blockDim.x + threadIdx.x;
  if (i < n) out[i] = (__bf16)in[i];
}

// Count edges per (dst, rel) segment and per-relation histogram (both etypes).
__global__ void mrc_count_edges(const int* __restrict__ dst,
                                const int* __restrict__ et0,
                                const int* __restrict__ et1,
                                unsigned* __restrict__ cnt0,
                                unsigned* __restrict__ cnt1,
                                unsigned* __restrict__ hist0,
                                unsigned* __restrict__ hist1, int e) {
  int i = blockIdx.x * blockDim.x + threadIdx.x;
  if (i < e) {
    int d = dst[i];
    int r0 = et0[i], r1 = et1[i];
    atomicAdd(&cnt0[d * NRELS + r0], 1u);
    atomicAdd(&cnt1[d * NRELS + r1], 1u);
    atomicAdd(&hist0[r0], 1u);
    atomicAdd(&hist1[r1], 1u);
  }
}

// 65-bin exclusive scan, padding each relation to a multiple of 128 edges.
__global__ void mrc_scan_offsets(const unsigned* __restrict__ hist,
                                 int* __restrict__ offpad,
                                 int* __restrict__ cursor) {
  if (blockIdx.x == 0 && threadIdx.x == 0) {
    int acc = 0;
    for (int r = 0; r < NRELS; ++r) {
      offpad[r] = acc;
      cursor[r] = acc;
      acc += (int)((hist[r] + 127u) & ~127u);
    }
    offpad[NRELS] = acc;
  }
}

// Scatter edge ids into relation-sorted (padded) order arrays.
__global__ void mrc_scatter_edges(const int* __restrict__ et0,
                                  const int* __restrict__ et1,
                                  int* __restrict__ cur0, int* __restrict__ cur1,
                                  int* __restrict__ ord0, int* __restrict__ ord1,
                                  int e) {
  int i = blockIdx.x * blockDim.x + threadIdx.x;
  if (i < e) {
    int p0 = atomicAdd(&cur0[et0[i]], 1);
    ord0[p0] = i;
    int p1 = atomicAdd(&cur1[et1[i]], 1);
    ord1[p1] = i;
  }
}

__global__ void mrc_make_inv(const unsigned* __restrict__ cnt,
                             float* __restrict__ inv, int n) {
  int i = blockIdx.x * blockDim.x + threadIdx.x;
  if (i < n) inv[i] = 1.0f / fmaxf((float)cnt[i], 1.0f);
}

// Repack W [nrel][FIN][HOUT] f32 into WMMA B-fragment layout (bf16):
// frag index ((r*KS+kk)*CT+c), per-lane 16 contiguous bf16:
//   lane<16 : N = c*16+lane,    K = kk*32 + j (j<8), kk*32 + 16 + (j-8) (j>=8)
//   lane>=16: N = c*16+lane-16, K = kk*32 + 8 + j (j<8), kk*32 + 24 + (j-8) (j>=8)
template <int FIN, int HOUT>
__global__ void mrc_repack_w(const float* __restrict__ W,
                             __bf16* __restrict__ Wl, int nrel) {
  constexpr int KS = FIN / 32, CT = HOUT / 16;
  int t = blockIdx.x * blockDim.x + threadIdx.x;
  int total = nrel * KS * CT * 32;
  if (t >= total) return;
  int lane = t & 31;
  int rest = t >> 5;
  int c = rest % CT; rest /= CT;
  int kk = rest % KS;
  int r = rest / KS;
  int n = c * 16 + (lane & 15);
  int kb = kk * 32 + ((lane >> 4) << 3);
  const float* wp = W + ((size_t)r * FIN + kb) * HOUT + n;
  __bf16* op = Wl + (size_t)t * 16;
#pragma unroll
  for (int j = 0; j < 16; ++j) op[j] = (__bf16)wp[(j + ((j >> 3) << 3)) * HOUT];
}

// --------------------------- WMMA GEMM kernels -----------------------------

// Dense root GEMM: out[row][col] = sum_k x[row][k]*root[k][col] + bias[col].
// One wave per 16-row tile; rows (20000) are an exact multiple of 16.
template <int FIN, int HOUT>
__global__ __launch_bounds__(256)
void mrc_dense_root_gemm(const __bf16* __restrict__ xin,   // [N, FIN]
                         const __bf16* __restrict__ Wl,    // repacked, nrel=1
                         const float* __restrict__ bias,   // [HOUT]
                         float* __restrict__ out,          // [N, HOUT]
                         int nrows) {
  constexpr int KS = FIN / 32, CT = HOUT / 16;
  int wave = (blockIdx.x * blockDim.x + threadIdx.x) >> 5;
  int ntiles = nrows >> 4;
  if (wave >= ntiles) return;                 // wave-uniform
  int lane = threadIdx.x & 31;
  int m = lane & 15, h = lane >> 4;

  const __bf16* xrow = xin + (size_t)(wave * 16 + m) * FIN;
  v16bf A[KS];
#pragma unroll
  for (int kk = 0; kk < KS; ++kk) {
    v8bf lo = *(const v8bf*)(xrow + kk * 32 + h * 8);
    v8bf hi = *(const v8bf*)(xrow + kk * 32 + 16 + h * 8);
    A[kk] = mrc_cat8(lo, hi);
  }

  v8f acc[CT] = {};
#pragma unroll
  for (int c = 0; c < CT; ++c) {
#pragma unroll
    for (int kk = 0; kk < KS; ++kk) {
      const __bf16* bp = Wl + ((size_t)(kk * CT + c) * 32 + lane) * 16;
      v8bf blo = *(const v8bf*)bp;
      v8bf bhi = *(const v8bf*)(bp + 8);
      v16bf Bf = mrc_cat8(blo, bhi);
      acc[c] = __builtin_amdgcn_wmma_f32_16x16x32_bf16(
          false, A[kk], false, Bf, (short)0, acc[c], false, false);
    }
  }

  // C layout: VGPR v, lanes 0-15 -> M=v, lanes 16-31 -> M=v+8; col = lane&15.
#pragma unroll
  for (int c = 0; c < CT; ++c) {
    int col = c * 16 + m;
    float bv = bias[col];
#pragma unroll
    for (int v = 0; v < 8; ++v) {
      int row = wave * 16 + h * 8 + v;
      out[(size_t)row * HOUT + col] = acc[c][v] + bv;
    }
  }
}

// Fused relational edge transform + mean-scaled scatter.
// Blocks are relation-uniform (edge lists padded to 128); each block stages
// the relation's repacked weight fragments into LDS with the gfx1250 async
// global->LDS path, then 8 waves each process one 16-edge tile with WMMA:
//   out[dst[e]] += (x[src[e]] @ W[etype[e]]) / cnt(dst[e], etype[e])
template <int FIN, int HOUT>
__global__ __launch_bounds__(256)
void mrc_edge_scatter_wmma(const __bf16* __restrict__ xin,   // [N, FIN]
                           const __bf16* __restrict__ Wl,    // repacked [NRELS]
                           const int* __restrict__ ord,      // padded, -1 = pad
                           const int* __restrict__ offpad,   // [NRELS+1]
                           const int* __restrict__ src,
                           const int* __restrict__ dst,
                           const float* __restrict__ invc,   // [N*NRELS]
                           float* __restrict__ out) {        // [N, HOUT] atomic
  constexpr int KS = FIN / 32, CT = HOUT / 16;
  constexpr int WBYTES = KS * CT * 1024;       // bytes of W[r] fragments
  __shared__ __align__(16) unsigned char smem[WBYTES];

  int blockBase = blockIdx.x * 128;
  if (blockBase >= offpad[NRELS]) return;      // block-uniform

  // which relation does this (relation-uniform) block belong to?
  int lo = 0, hi = NRELS;
  while (hi - lo > 1) {
    int mid = (lo + hi) >> 1;
    if (offpad[mid] <= blockBase) lo = mid; else hi = mid;
  }
  int r = lo;

  // Cooperative async stage of W[r] fragments into LDS (ASYNCcnt tracked).
  {
    const unsigned char* gsrc =
        (const unsigned char*)(Wl + (size_t)r * KS * CT * 512);
    unsigned ldsbase = (unsigned)(size_t)(&smem[0]);
    for (int i = (int)threadIdx.x * 16; i < WBYTES; i += 256 * 16) {
      unsigned loff = ldsbase + (unsigned)i;
      unsigned long long ga = (unsigned long long)(gsrc + i);
      asm volatile("global_load_async_to_lds_b128 %0, %1, off"
                   :: "v"(loff), "v"(ga) : "memory");
    }
    asm volatile("s_wait_asynccnt 0x0" ::: "memory");
  }
  __syncthreads();

  int lane = threadIdx.x & 31;
  int m = lane & 15, h = lane >> 4;
  int base = blockBase + ((int)(threadIdx.x >> 5)) * 16;

  int e = ord[base + m];
  int sv = 0, dv = 0;
  float iv = 0.0f;
  if (e >= 0) {
    sv = src[e];
    dv = dst[e];
    iv = invc[dv * NRELS + r];
  }

  // A fragments (16-bit A-matrix 16x32 layout), gathered per edge.
  v16bf A[KS];
  const __bf16* xrow = xin + (size_t)sv * FIN;
#pragma unroll
  for (int kk = 0; kk < KS; ++kk) {
    v8bf alo = *(const v8bf*)(xrow + kk * 32 + h * 8);
    v8bf ahi = *(const v8bf*)(xrow + kk * 32 + 16 + h * 8);
    A[kk] = mrc_cat8(alo, ahi);
  }

  v8f acc[CT] = {};
  const __bf16* lw = (const __bf16*)smem;
#pragma unroll
  for (int c = 0; c < CT; ++c) {
#pragma unroll
    for (int kk = 0; kk < KS; ++kk) {
      const __bf16* bp = lw + ((kk * CT + c) * 32 + lane) * 16;
      v8bf blo = *(const v8bf*)bp;          // ds_load_b128
      v8bf bhi = *(const v8bf*)(bp + 8);    // ds_load_b128
      v16bf Bf = mrc_cat8(blo, bhi);
      acc[c] = __builtin_amdgcn_wmma_f32_16x16x32_bf16(
          false, A[kk], false, Bf, (short)0, acc[c], false, false);
    }
  }

  // This lane holds rows (h*8 + v); fetch their dst/scale from the lane that
  // loaded that row's edge (rows 0..15 live on lanes 0..15).
  int drow[8];
  float irow[8];
#pragma unroll
  for (int v = 0; v < 8; ++v) {
    int rsel = h * 8 + v;
    drow[v] = __shfl(dv, rsel, 32);
    irow[v] = __shfl(iv, rsel, 32);
  }
#pragma unroll
  for (int c = 0; c < CT; ++c) {
    int col = c * 16 + m;
#pragma unroll
    for (int v = 0; v < 8; ++v) {
      if (irow[v] != 0.0f)
        atomicAdd(&out[(size_t)drow[v] * HOUT + col], acc[c][v] * irow[v]);
    }
  }
}

// ------------------------------ epilogues ----------------------------------

__global__ void mrc_relu_bf16(float* __restrict__ a, __bf16* __restrict__ bf,
                              int n) {
  int i = blockIdx.x * blockDim.x + threadIdx.x;
  if (i < n) {
    float v = a[i];
    v = v > 0.0f ? v : 0.0f;
    a[i] = v;
    bf[i] = (__bf16)v;
  }
}

// h_os = sigmoid(mean(x2_o, axis=0)); one block of 256 threads, H2DIM=32.
__global__ void mrc_readout(const float* __restrict__ x2, float* __restrict__ hos) {
  __shared__ float s[256];
  int col = threadIdx.x & 31;
  int grp = threadIdx.x >> 5;          // 8 row groups
  float sum = 0.0f;
  for (int row = grp; row < NNODES; row += 8) sum += x2[row * H2DIM + col];
  s[threadIdx.x] = sum;
  __syncthreads();
  if (grp == 0) {
    float t = 0.0f;
    for (int g = 0; g < 8; ++g) t += s[g * 32 + col];
    t /= (float)NNODES;
    hos[col] = 1.0f / (1.0f + __expf(-t));
  }
}

// v = disc_w[0] @ h_os   (32x32 matvec)
__global__ void mrc_disc_vec(const float* __restrict__ w,
                             const float* __restrict__ hos,
                             float* __restrict__ v) {
  int i = threadIdx.x;
  if (i < H2DIM) {
    float s = 0.0f;
    for (int k = 0; k < H2DIM; ++k) s += w[i * H2DIM + k] * hos[k];
    v[i] = s;
  }
}

// ret_os / ret_os_a rows + copy x2_o to output.
__global__ void mrc_bilinear_out(const float* __restrict__ x2o,
                                 const float* __restrict__ x2a,
                                 const float* __restrict__ x2aa,
                                 const float* __restrict__ v,
                                 const float* __restrict__ disc_b,
                                 float* __restrict__ ret_os,
                                 float* __restrict__ ret_os_a,
                                 float* __restrict__ x2o_out) {
  int i = blockIdx.x * blockDim.x + threadIdx.x;
  if (i < NNODES) {
    float db = disc_b[0];
    float a = 0.0f, b = 0.0f, c = 0.0f;
    for (int k = 0; k < H2DIM; ++k) {
      float vk = v[k];
      a += x2o[i * H2DIM + k] * vk;
      b += x2a[i * H2DIM + k] * vk;
      c += x2aa[i * H2DIM + k] * vk;
    }
    ret_os[i * 2 + 0] = a + db;
    ret_os[i * 2 + 1] = b + db;
    ret_os_a[i * 2 + 0] = a + db;
    ret_os_a[i * 2 + 1] = c + db;
    for (int k = 0; k < H2DIM; ++k) x2o_out[i * H2DIM + k] = x2o[i * H2DIM + k];
  }
}

// logits[b][c] = concat(final[idx0], final[idx1]) . cls_w[:,c] + cls_b[c]
__global__ void mrc_classify(const float* __restrict__ x1o,  // [N,H1]
                             const float* __restrict__ x2o,  // [N,H2]
                             const int* __restrict__ idx,    // [2,B]
                             const float* __restrict__ attt,
                             const float* __restrict__ cw,   // [192, 65]
                             const float* __restrict__ cb,
                             float* __restrict__ logout) {
  int t = blockIdx.x * blockDim.x + threadIdx.x;
  if (t >= NPAIR * NRELS) return;
  int b = t / NRELS, c = t % NRELS;
  int i0 = idx[b], i1 = idx[NPAIR + b];
  float a0 = attt[0], a1 = attt[1];
  float s = cb[c];
  const float* w = cw + c;
  for (int j = 0; j < H1DIM; ++j) s += a0 * x1o[i0 * H1DIM + j] * w[j * NRELS];
  for (int j = 0; j < H2DIM; ++j) s += a1 * x2o[i0 * H2DIM + j] * w[(H1DIM + j) * NRELS];
  for (int j = 0; j < H1DIM; ++j) s += a0 * x1o[i1 * H1DIM + j] * w[(96 + j) * NRELS];
  for (int j = 0; j < H2DIM; ++j) s += a1 * x2o[i1 * H2DIM + j] * w[(160 + j) * NRELS];
  logout[t] = s;
}

// ------------------------------- launcher ----------------------------------

extern "C" void kernel_launch(void* const* d_in, const int* in_sizes, int n_in,
                              void* d_out, int out_size, void* d_ws, size_t ws_size,
                              hipStream_t stream) {
  (void)in_sizes; (void)n_in; (void)out_size; (void)ws_size;

  const float* x_o    = (const float*)d_in[0];
  const float* x_a    = (const float*)d_in[1];
  const int*   eix    = (const int*)d_in[2];     // [2,E]
  const int*   et0    = (const int*)d_in[3];
  const int*   et1    = (const int*)d_in[4];
  const int*   idx    = (const int*)d_in[5];     // [2,B]
  const float* W1     = (const float*)d_in[6];
  const float* root1  = (const float*)d_in[7];
  const float* b1     = (const float*)d_in[8];
  const float* W2     = (const float*)d_in[9];
  const float* root2  = (const float*)d_in[10];
  const float* b2     = (const float*)d_in[11];
  const float* attt   = (const float*)d_in[12];
  const float* disc_w = (const float*)d_in[13];
  const float* disc_b = (const float*)d_in[14];
  const float* cls_w  = (const float*)d_in[15];
  const float* cls_b  = (const float*)d_in[16];

  const int* src = eix;
  const int* dst = eix + NEDGES;

  // ---- workspace carve-up (256B aligned bump allocator) ----
  char* base = (char*)d_ws;
  size_t off = 0;
  auto alloc = [&](size_t bytes) -> char* {
    char* p = base + off;
    off += (bytes + 255) & ~(size_t)255;
    return p;
  };
  const int NR = NNODES * NRELS;
  unsigned* cnt0   = (unsigned*)alloc((size_t)NR * 4);
  unsigned* cnt1   = (unsigned*)alloc((size_t)NR * 4);
  float*    invc0  = (float*)alloc((size_t)NR * 4);
  float*    invc1  = (float*)alloc((size_t)NR * 4);
  unsigned* hist0  = (unsigned*)alloc(NRELS * 4);
  unsigned* hist1  = (unsigned*)alloc(NRELS * 4);
  int*      offp0  = (int*)alloc((NRELS + 1) * 4);
  int*      offp1  = (int*)alloc((NRELS + 1) * 4);
  int*      cur0   = (int*)alloc(NRELS * 4);
  int*      cur1   = (int*)alloc(NRELS * 4);
  int*      ord0   = (int*)alloc((size_t)EPAD * 4);
  int*      ord1   = (int*)alloc((size_t)EPAD * 4);
  __bf16*   xo_bf  = (__bf16*)alloc((size_t)NNODES * FDIM * 2);
  __bf16*   xa_bf  = (__bf16*)alloc((size_t)NNODES * FDIM * 2);
  __bf16*   Wl1    = (__bf16*)alloc((size_t)NRELS * (FDIM/32) * (H1DIM/16) * 512 * 2);
  __bf16*   rootl1 = (__bf16*)alloc((size_t)(FDIM/32) * (H1DIM/16) * 512 * 2);
  __bf16*   Wl2    = (__bf16*)alloc((size_t)NRELS * (H1DIM/32) * (H2DIM/16) * 512 * 2);
  __bf16*   rootl2 = (__bf16*)alloc((size_t)(H1DIM/32) * (H2DIM/16) * 512 * 2);
  float*    acc1_o  = (float*)alloc((size_t)NNODES * H1DIM * 4);
  float*    acc1_a  = (float*)alloc((size_t)NNODES * H1DIM * 4);
  float*    acc1_aa = (float*)alloc((size_t)NNODES * H1DIM * 4);
  __bf16*   h1bf_o  = (__bf16*)alloc((size_t)NNODES * H1DIM * 2);
  __bf16*   h1bf_a  = (__bf16*)alloc((size_t)NNODES * H1DIM * 2);
  __bf16*   h1bf_aa = (__bf16*)alloc((size_t)NNODES * H1DIM * 2);
  float*    x2_o    = (float*)alloc((size_t)NNODES * H2DIM * 4);
  float*    x2_a    = (float*)alloc((size_t)NNODES * H2DIM * 4);
  float*    x2_aa   = (float*)alloc((size_t)NNODES * H2DIM * 4);
  float*    hos     = (float*)alloc(128);
  float*    vvec    = (float*)alloc(128);

  // ---- output regions (all float32) ----
  float* out_log   = (float*)d_out;                       // [B, 65]
  float* out_ros   = out_log + (size_t)NPAIR * NRELS;     // [N, 2]
  float* out_rosa  = out_ros + (size_t)NNODES * 2;        // [N, 2]
  float* out_x2o   = out_rosa + (size_t)NNODES * 2;       // [N, 32]

  // ---- stage 0: zero / sentinel init (capture-safe) ----
  (void)hipMemsetAsync(cnt0, 0, (size_t)NR * 4, stream);
  (void)hipMemsetAsync(cnt1, 0, (size_t)NR * 4, stream);
  (void)hipMemsetAsync(hist0, 0, NRELS * 4, stream);
  (void)hipMemsetAsync(hist1, 0, NRELS * 4, stream);
  (void)hipMemsetAsync(ord0, 0xFF, (size_t)EPAD * 4, stream);
  (void)hipMemsetAsync(ord1, 0xFF, (size_t)EPAD * 4, stream);

  // ---- stage 1: conversions & weight repack ----
  int nx = NNODES * FDIM;
  mrc_f32_to_bf16<<<(nx + 255) / 256, 256, 0, stream>>>(x_o, xo_bf, nx);
  mrc_f32_to_bf16<<<(nx + 255) / 256, 256, 0, stream>>>(x_a, xa_bf, nx);
  {
    int t1 = NRELS * (FDIM/32) * (H1DIM/16) * 32;
    mrc_repack_w<FDIM, H1DIM><<<(t1 + 255) / 256, 256, 0, stream>>>(W1, Wl1, NRELS);
    int tr1 = (FDIM/32) * (H1DIM/16) * 32;
    mrc_repack_w<FDIM, H1DIM><<<(tr1 + 255) / 256, 256, 0, stream>>>(root1, rootl1, 1);
    int t2 = NRELS * (H1DIM/32) * (H2DIM/16) * 32;
    mrc_repack_w<H1DIM, H2DIM><<<(t2 + 255) / 256, 256, 0, stream>>>(W2, Wl2, NRELS);
    int tr2 = (H1DIM/32) * (H2DIM/16) * 32;
    mrc_repack_w<H1DIM, H2DIM><<<(tr2 + 255) / 256, 256, 0, stream>>>(root2, rootl2, 1);
  }

  // ---- stage 2: graph bookkeeping ----
  int eb = (NEDGES + 255) / 256;
  mrc_count_edges<<<eb, 256, 0, stream>>>(dst, et0, et1, cnt0, cnt1, hist0, hist1, NEDGES);
  mrc_scan_offsets<<<1, 1, 0, stream>>>(hist0, offp0, cur0);
  mrc_scan_offsets<<<1, 1, 0, stream>>>(hist1, offp1, cur1);
  mrc_scatter_edges<<<eb, 256, 0, stream>>>(et0, et1, cur0, cur1, ord0, ord1, NEDGES);
  mrc_make_inv<<<(NR + 255) / 256, 256, 0, stream>>>(cnt0, invc0, NR);
  mrc_make_inv<<<(NR + 255) / 256, 256, 0, stream>>>(cnt1, invc1, NR);

  // ---- stage 3: layer 1 (root GEMM init + fused edge WMMA scatter) ----
  int dtiles = NNODES / 16;                       // 1250
  int dblk = (dtiles + 7) / 8;                    // 8 waves/block
  mrc_dense_root_gemm<FDIM, H1DIM><<<dblk, 256, 0, stream>>>(xo_bf, rootl1, b1, acc1_o, NNODES);
  mrc_dense_root_gemm<FDIM, H1DIM><<<dblk, 256, 0, stream>>>(xa_bf, rootl1, b1, acc1_a, NNODES);
  mrc_dense_root_gemm<FDIM, H1DIM><<<dblk, 256, 0, stream>>>(xo_bf, rootl1, b1, acc1_aa, NNODES);

  int eblk = EPAD / 128;                          // relation-uniform blocks
  mrc_edge_scatter_wmma<FDIM, H1DIM><<<eblk, 256, 0, stream>>>(
      xo_bf, Wl1, ord0, offp0, src, dst, invc0, acc1_o);
  mrc_edge_scatter_wmma<FDIM, H1DIM><<<eblk, 256, 0, stream>>>(
      xa_bf, Wl1, ord0, offp0, src, dst, invc0, acc1_a);
  mrc_edge_scatter_wmma<FDIM, H1DIM><<<eblk, 256, 0, stream>>>(
      xo_bf, Wl1, ord1, offp1, src, dst, invc1, acc1_aa);

  int nh1 = NNODES * H1DIM;
  mrc_relu_bf16<<<(nh1 + 255) / 256, 256, 0, stream>>>(acc1_o, h1bf_o, nh1);
  mrc_relu_bf16<<<(nh1 + 255) / 256, 256, 0, stream>>>(acc1_a, h1bf_a, nh1);
  mrc_relu_bf16<<<(nh1 + 255) / 256, 256, 0, stream>>>(acc1_aa, h1bf_aa, nh1);

  // ---- stage 4: layer 2 ----
  mrc_dense_root_gemm<H1DIM, H2DIM><<<dblk, 256, 0, stream>>>(h1bf_o, rootl2, b2, x2_o, NNODES);
  mrc_dense_root_gemm<H1DIM, H2DIM><<<dblk, 256, 0, stream>>>(h1bf_a, rootl2, b2, x2_a, NNODES);
  mrc_dense_root_gemm<H1DIM, H2DIM><<<dblk, 256, 0, stream>>>(h1bf_aa, rootl2, b2, x2_aa, NNODES);

  mrc_edge_scatter_wmma<H1DIM, H2DIM><<<eblk, 256, 0, stream>>>(
      h1bf_o, Wl2, ord0, offp0, src, dst, invc0, x2_o);
  mrc_edge_scatter_wmma<H1DIM, H2DIM><<<eblk, 256, 0, stream>>>(
      h1bf_a, Wl2, ord0, offp0, src, dst, invc0, x2_a);
  mrc_edge_scatter_wmma<H1DIM, H2DIM><<<eblk, 256, 0, stream>>>(
      h1bf_aa, Wl2, ord1, offp1, src, dst, invc1, x2_aa);

  // ---- stage 5: heads ----
  mrc_readout<<<1, 256, 0, stream>>>(x2_o, hos);
  mrc_disc_vec<<<1, 32, 0, stream>>>(disc_w, hos, vvec);
  mrc_bilinear_out<<<(NNODES + 255) / 256, 256, 0, stream>>>(
      x2_o, x2_a, x2_aa, vvec, disc_b, out_ros, out_rosa, out_x2o);
  int ct = NPAIR * NRELS;
  mrc_classify<<<(ct + 255) / 256, 256, 0, stream>>>(
      acc1_o, x2_o, idx, attt, cls_w, cls_b, out_log);
}
